// VimSSM_35330400977099
// MI455X (gfx1250) — matrix operand order repaired and checked by reading, not compiled, not run on hardware
//
#include <hip/hip_runtime.h>
#include <hip/hip_bf16.h>

typedef __attribute__((ext_vector_type(16))) _Float16 v16h;
typedef __attribute__((ext_vector_type(8)))  _Float16 v8h;
typedef __attribute__((ext_vector_type(8)))  float    v8f;

namespace {
constexpr int D_MODEL = 768;
constexpr int D_STATE = 16;
constexpr int D_INNER = 1536;           // 768 * 2
constexpr int BATCH   = 2;
constexpr int SEQ     = 2048;
constexpr int M_ROWS  = BATCH * SEQ;    // 4096 token rows
constexpr int N_DBL   = 2 * D_STATE + D_INNER;  // 1568
}

// ---------------------------------------------------------------------------
// fp32 -> fp16 elementwise convert (activations / weights for WMMA operands)
// ---------------------------------------------------------------------------
__global__ void cvt_f32_to_f16(const float* __restrict__ in,
                               _Float16* __restrict__ out, int n) {
    int i = blockIdx.x * blockDim.x + threadIdx.x;
    if (i < n) out[i] = (_Float16)in[i];
}

// ---------------------------------------------------------------------------
// WMMA GEMM:  C[m][n] = sum_k A[m][k] * W[n][k]   (A: MxK f16, W: NxK f16)
// Block = 128 threads (4 waves). Each wave -> 16(M) x NTILES*16(N) strip:
// one A tile (16x32) reused across NTILES B tiles, v_wmma_f32_16x16x32_f16.
// NTILES is a compile-time constant so the k-loop is completely branch-free
// (keeps EXEC all-ones at the WMMA and lets accumulators stay in place).
// Optional f16 mirror of columns < auxN feeds the next GEMM.
// ---------------------------------------------------------------------------
template <int NTILES>
__global__ __launch_bounds__(128)
void gemm_wmma_f16(const _Float16* __restrict__ A,
                   const _Float16* __restrict__ W,
                   float* __restrict__ C,
                   _Float16* __restrict__ auxC,
                   int M, int N, int K, int auxN, int col0) {
    const int lane     = threadIdx.x & 31;
    const int wave     = threadIdx.x >> 5;
    const int row_base = blockIdx.y * 64 + wave * 16;          // M % 64 == 0
    const int col_base = col0 + blockIdx.x * (NTILES * 16);

    // A operand (16-bit A 16x32): lanes 0-15 -> M=lane, K in {0..15};
    // lanes 16-31 -> same M, K in {16..31}.
    const int am  = lane & 15;
    const int akh = (lane >> 4) * 16;
    // B operand (16-bit B 32x16): lane&15 -> N column; half-wave selects
    // K range 0..15 vs 16..31; each lane holds 16 contiguous K values.
    const int bn  = lane & 15;
    const int bkh = (lane >> 4) * 16;

    const _Float16* pa = A + (size_t)(row_base + am) * K + akh;
    const _Float16* pb0 = W + (size_t)(col_base + bn) * K + bkh;

    v8f acc0 = {}, acc1 = {}, acc2 = {}, acc3 = {};

    for (int k0 = 0; k0 < K; k0 += 32) {
        v8h alo = *(const v8h*)(pa + k0);
        v8h ahi = *(const v8h*)(pa + k0 + 8);
        v16h a  = __builtin_shufflevector(alo, ahi,
                      0,1,2,3,4,5,6,7,8,9,10,11,12,13,14,15);
        __builtin_prefetch(pa + k0 + 64, 0, 3);   // speculative, branch-free

        {
            v16h b = *(const v16h*)(pb0 + k0);
            acc0 = __builtin_amdgcn_wmma_f32_16x16x32_f16(
                false, a, false, b, (short)0, acc0, false, false);
        }
        if constexpr (NTILES > 1) {
            v16h b = *(const v16h*)(pb0 + (size_t)16 * K + k0);
            acc1 = __builtin_amdgcn_wmma_f32_16x16x32_f16(
                false, a, false, b, (short)0, acc1, false, false);
        }
        if constexpr (NTILES > 2) {
            v16h b = *(const v16h*)(pb0 + (size_t)32 * K + k0);
            acc2 = __builtin_amdgcn_wmma_f32_16x16x32_f16(
                false, a, false, b, (short)0, acc2, false, false);
        }
        if constexpr (NTILES > 3) {
            v16h b = *(const v16h*)(pb0 + (size_t)48 * K + k0);
            acc3 = __builtin_amdgcn_wmma_f32_16x16x32_f16(
                false, a, false, b, (short)0, acc3, false, false);
        }
    }

    // C/D layout: VGPR r -> M = r + 8*(lane>=16), N = lane&15
    const int r0 = row_base + ((lane >> 4) * 8);
    const v8f accs[4] = { acc0, acc1, acc2, acc3 };
#pragma unroll
    for (int j = 0; j < NTILES; ++j) {
        const int col = col_base + j * 16 + (lane & 15);
#pragma unroll
        for (int r = 0; r < 8; ++r) {
            float v = accs[j][r];
            C[(size_t)(r0 + r) * N + col] = v;
            if (auxC && col < auxN)
                auxC[(size_t)(r0 + r) * auxN + col] = (_Float16)v;
        }
    }
}

// ---------------------------------------------------------------------------
// Depthwise causal conv1d (k=4) + bias + SiLU. Reads x_in = xz[..., :D_INNER].
// Writes x_conv (f32, for the scan) and an f16 mirror (GEMM2 A operand).
// ---------------------------------------------------------------------------
__global__ void conv_silu_kernel(const float* __restrict__ xz,
                                 const float* __restrict__ conv_w,
                                 const float* __restrict__ conv_b,
                                 float* __restrict__ x_conv,
                                 _Float16* __restrict__ x_conv_h) {
    int idx = blockIdx.x * blockDim.x + threadIdx.x;   // (b*L + t)*D_INNER + d
    if (idx >= BATCH * SEQ * D_INNER) return;
    int d = idx % D_INNER;
    int t = (idx / D_INNER) % SEQ;
    int b = idx / (D_INNER * SEQ);
    float s = conv_b[d];
#pragma unroll
    for (int j = 0; j < 4; ++j) {
        int tt = t + j - 3;
        if (tt >= 0)
            s += xz[((size_t)(b * SEQ + tt)) * (2 * D_INNER) + d] * conv_w[d * 4 + j];
    }
    float r = s / (1.f + __expf(-s));                  // SiLU
    x_conv[idx]   = r;
    x_conv_h[idx] = (_Float16)r;
}

// ---------------------------------------------------------------------------
// Sequential selective scan. One lane per (batch, channel); 16 states in
// registers. Fuses: softplus(delta_lin + dt_b), recurrence, y = h.C + u*D,
// gating by silu(z), and the f16 cast feeding the out_proj GEMM.
// ---------------------------------------------------------------------------
__global__ __launch_bounds__(256)
void scan_kernel(const float* __restrict__ delta_lin,
                 const float* __restrict__ x_dbl,
                 const float* __restrict__ x_conv,
                 const float* __restrict__ xz,
                 const float* __restrict__ A_log,
                 const float* __restrict__ D_param,
                 const float* __restrict__ dt_b,
                 _Float16* __restrict__ y_h) {
    int ch = blockIdx.x * blockDim.x + threadIdx.x;    // 0 .. BATCH*D_INNER-1
    if (ch >= BATCH * D_INNER) return;
    int b = ch / D_INNER;
    int d = ch - b * D_INNER;

    float An[D_STATE];
#pragma unroll
    for (int n = 0; n < D_STATE; ++n) An[n] = -__expf(A_log[d * D_STATE + n]);
    const float Dp    = D_param[d];
    const float dbias = dt_b[d];

    float h[D_STATE];
#pragma unroll
    for (int n = 0; n < D_STATE; ++n) h[n] = 0.f;

    for (int t = 0; t < SEQ; ++t) {
        size_t row = (size_t)b * SEQ + t;
        float dr    = delta_lin[row * D_INNER + d] + dbias;
        float delta = (dr > 20.f) ? dr : log1pf(__expf(dr));   // softplus
        float u     = x_conv[row * D_INNER + d];
        float zv    = xz[row * (2 * D_INNER) + D_INNER + d];

        const float* bc = x_dbl + row * (size_t)N_DBL + D_INNER;  // lane-uniform
        float Bv[D_STATE], Cv[D_STATE];
#pragma unroll
        for (int n = 0; n < D_STATE; ++n) { Bv[n] = bc[n]; Cv[n] = bc[D_STATE + n]; }

        float du = delta * u;
        float y  = 0.f;
#pragma unroll
        for (int n = 0; n < D_STATE; ++n) {
            h[n] = __expf(delta * An[n]) * h[n] + du * Bv[n];
            y   += h[n] * Cv[n];
        }
        y += u * Dp;
        y *= zv / (1.f + __expf(-zv));                 // * silu(z)
        y_h[row * D_INNER + d] = (_Float16)y;
    }
}

// ---------------------------------------------------------------------------
extern "C" void kernel_launch(void* const* d_in, const int* in_sizes, int n_in,
                              void* d_out, int out_size, void* d_ws, size_t ws_size,
                              hipStream_t stream) {
    (void)in_sizes; (void)n_in; (void)out_size; (void)ws_size;
    const float* x          = (const float*)d_in[0];   // (B,L,768)
    const float* in_proj_w  = (const float*)d_in[1];   // (3072,768)
    const float* conv_w     = (const float*)d_in[2];   // (1536,1,4)
    const float* conv_b     = (const float*)d_in[3];   // (1536,)
    const float* x_proj_w   = (const float*)d_in[4];   // (1568,1536)
    const float* dt_proj_w  = (const float*)d_in[5];   // (1536,1536)
    const float* dt_proj_b  = (const float*)d_in[6];   // (1536,)
    const float* A_log      = (const float*)d_in[7];   // (1536,16)
    const float* D_param    = (const float*)d_in[8];   // (1536,)
    const float* out_proj_w = (const float*)d_in[9];   // (768,1536)
    float* out = (float*)d_out;                        // (B,L,768) f32

    // ---- workspace layout (256B aligned carve-outs) ----
    char*  base = (char*)d_ws;
    size_t off  = 0;
    auto take = [&](size_t bytes) -> void* {
        void* p = base + off;
        off += (bytes + 255) & ~(size_t)255;
        return p;
    };
    _Float16* x_h         = (_Float16*)take((size_t)M_ROWS * D_MODEL * 2);
    _Float16* w_in_h      = (_Float16*)take((size_t)2 * D_INNER * D_MODEL * 2);
    _Float16* w_xproj_h   = (_Float16*)take((size_t)N_DBL * D_INNER * 2);
    _Float16* w_dt_h      = (_Float16*)take((size_t)D_INNER * D_INNER * 2);
    _Float16* w_out_h     = (_Float16*)take((size_t)D_MODEL * D_INNER * 2);
    float*    xz          = (float*)   take((size_t)M_ROWS * 2 * D_INNER * 4);
    float*    x_conv      = (float*)   take((size_t)M_ROWS * D_INNER * 4);
    _Float16* x_conv_h    = (_Float16*)take((size_t)M_ROWS * D_INNER * 2);
    float*    x_dbl       = (float*)   take((size_t)M_ROWS * N_DBL * 4);
    _Float16* delta_raw_h = (_Float16*)take((size_t)M_ROWS * D_INNER * 2);
    float*    delta_lin   = (float*)   take((size_t)M_ROWS * D_INNER * 4);
    _Float16* y_h         = (_Float16*)take((size_t)M_ROWS * D_INNER * 2);

    auto cvt = [&](const float* src, _Float16* dst, int n) {
        cvt_f32_to_f16<<<(n + 255) / 256, 256, 0, stream>>>(src, dst, n);
    };
    // f16 operand prep
    cvt(x,          x_h,       M_ROWS * D_MODEL);
    cvt(in_proj_w,  w_in_h,    2 * D_INNER * D_MODEL);
    cvt(x_proj_w,   w_xproj_h, N_DBL * D_INNER);
    cvt(dt_proj_w,  w_dt_h,    D_INNER * D_INNER);
    cvt(out_proj_w, w_out_h,   D_MODEL * D_INNER);

    // GEMM launcher: full 64-col blocks with NTILES=4; N=1568 leaves a 32-col
    // remainder which goes out as a separate branch-free NTILES=2 launch.
    auto gemm = [&](const _Float16* A, const _Float16* W, float* C,
                    _Float16* aux, int M, int N, int K, int auxN) {
        int full = N / 64;
        if (full > 0) {
            dim3 grid(full, M / 64);
            gemm_wmma_f16<4><<<grid, 128, 0, stream>>>(A, W, C, aux,
                                                       M, N, K, auxN, 0);
        }
        int rem = (N - full * 64) / 16;     // 0 or 2 for our shapes
        if (rem == 2) {
            dim3 grid(1, M / 64);
            gemm_wmma_f16<2><<<grid, 128, 0, stream>>>(A, W, C, aux,
                                                       M, N, K, auxN, full * 64);
        }
    };

    // 1) xz = x @ in_proj_w^T            (4096 x 3072)
    gemm(x_h, w_in_h, xz, nullptr, M_ROWS, 2 * D_INNER, D_MODEL, 0);

    // 2) depthwise conv + SiLU
    {
        int n = M_ROWS * D_INNER;
        conv_silu_kernel<<<(n + 255) / 256, 256, 0, stream>>>(
            xz, conv_w, conv_b, x_conv, x_conv_h);
    }

    // 3) x_dbl = x_conv @ x_proj_w^T     (4096 x 1568), mirror delta_raw as f16
    gemm(x_conv_h, w_xproj_h, x_dbl, delta_raw_h, M_ROWS, N_DBL, D_INNER, D_INNER);

    // 4) delta_lin = delta_raw @ dt_proj_w^T   (4096 x 1536)
    gemm(delta_raw_h, w_dt_h, delta_lin, nullptr, M_ROWS, D_INNER, D_INNER, 0);

    // 5) sequential selective scan + gating -> y_h
    {
        int n = BATCH * D_INNER;           // 3072 channels
        scan_kernel<<<(n + 255) / 256, 256, 0, stream>>>(
            delta_lin, x_dbl, x_conv, xz, A_log, D_param, dt_proj_b, y_h);
    }

    // 6) out = y @ out_proj_w^T          (4096 x 768)
    gemm(y_h, w_out_h, out, nullptr, M_ROWS, D_MODEL, D_INNER, 0);
}
